// Attn5Encoder_7413113553590
// MI455X (gfx1250) — matrix operand order repaired and checked by reading, not compiled, not run on hardware
//
#include <hip/hip_runtime.h>
#include <hip/hip_bf16.h>

// ---------------------------------------------------------------------------
// Attn5Encoder for MI455X (gfx1250, wave32, WMMA).
// Pipeline: embed -> gi GEMMs -> BiGRU scan (L0) -> gi GEMMs -> BiGRU scan (L1)
//           -> wh/wh2 GEMMs -> fwh/vwh energy GEMMs -> softmax -> transpose
//           -> context GEMM.  All GEMMs use v_wmma_f32_16x16x32_bf16.
// GEMM tile staging uses GLOBAL_LOAD_ASYNC_TO_LDS_B128 (ASYNCcnt) via inline
// asm (ISA operand order: lds-dst VGPR, 64-bit global vaddr, off).
// ---------------------------------------------------------------------------

typedef __attribute__((ext_vector_type(16))) __bf16 v16bf;
typedef __attribute__((ext_vector_type(8)))  float  v8f;

#define B_   32
#define L_   512
#define E_   256
#define H_   256
#define H2_  512
#define G3_  768          // 3*H
#define F_   512          // NUM_FEATURES
#define NF_  384
#define PAD_ 128

static __device__ __forceinline__ v8f wmma_bf16(v16bf a, v16bf b, v8f c) {
  return __builtin_amdgcn_wmma_f32_16x16x32_bf16(false, a, false, b, (short)0, c,
                                                 false, false);
}

// async copy 16B: global -> LDS, tracked by ASYNCcnt
static __device__ __forceinline__ void async_g2l_b128(void* lds, const void* g) {
  unsigned lds_off = (unsigned)(size_t)lds;   // LDS aperture: addr[31:0]
  asm volatile("global_load_async_to_lds_b128 %0, %1, off"
               :
               : "v"(lds_off), "v"(g)
               : "memory");
}

static __device__ __forceinline__ void wait_asynccnt0() {
  asm volatile("s_wait_asynccnt 0x0" ::: "memory");
}

// ------------------------------- small kernels -----------------------------

__global__ void k_f32_to_bf16(const float* __restrict__ src,
                              __bf16* __restrict__ dst, long n) {
  long i = (long)blockIdx.x * blockDim.x + threadIdx.x;
  if (i < n) dst[i] = (__bf16)src[i];
}

// x_bf[(l*B+b)*E + e] = emb_table[tok[b*L+l]*E + e]
__global__ void k_embed(const int* __restrict__ tok,
                        const float* __restrict__ table,
                        __bf16* __restrict__ xb) {
  long i = (long)blockIdx.x * blockDim.x + threadIdx.x;     // L*B*E
  if (i >= (long)L_ * B_ * E_) return;
  int e  = (int)(i & (E_ - 1));
  long lb = i >> 8;
  int b  = (int)(lb & (B_ - 1));
  int l  = (int)(lb >> 5);
  int t  = tok[(long)b * L_ + l];
  xb[i] = (__bf16)table[(long)t * E_ + e];
}

// rnnT[b][j][l] = rnn_bf[(l*B+b)*H2 + j]     (per-batch transpose for context)
__global__ void k_transpose_rnn(const __bf16* __restrict__ rnn,
                                __bf16* __restrict__ rnnT) {
  long i = (long)blockIdx.x * blockDim.x + threadIdx.x;     // B*H2*L
  if (i >= (long)B_ * H2_ * L_) return;
  int l  = (int)(i & (L_ - 1));
  long bj = i >> 9;
  int j  = (int)(bj & (H2_ - 1));
  int b  = (int)(bj >> 9);
  rnnT[i] = rnn[((long)l * B_ + b) * H2_ + j];
}

// softmax over F for energy (B,L,F); writes weights (B,F,L) f32 + bf16
__global__ __launch_bounds__(256)
void k_softmax(const float* __restrict__ energy, float* __restrict__ w_out,
               __bf16* __restrict__ w_bf) {
  __shared__ float red[256];
  int bl = blockIdx.x;                  // b*L + l
  int b  = bl >> 9, l = bl & (L_ - 1);
  const float* e = energy + (long)bl * F_;
  int t = threadIdx.x;
  float v0 = e[t], v1 = e[t + 256];
  red[t] = fmaxf(v0, v1);
  __syncthreads();
  for (int s = 128; s > 0; s >>= 1) {
    if (t < s) red[t] = fmaxf(red[t], red[t + s]);
    __syncthreads();
  }
  float mx = red[0];
  __syncthreads();
  float e0 = expf(v0 - mx), e1 = expf(v1 - mx);
  red[t] = e0 + e1;
  __syncthreads();
  for (int s = 128; s > 0; s >>= 1) {
    if (t < s) red[t] += red[t + s];
    __syncthreads();
  }
  float inv = 1.0f / red[0];
  long base = (long)b * F_ * L_ + l;
  float r0 = e0 * inv, r1 = e1 * inv;
  w_out[base + (long)t * L_]         = r0;
  w_out[base + (long)(t + 256) * L_] = r1;
  w_bf[base + (long)t * L_]          = (__bf16)r0;
  w_bf[base + (long)(t + 256) * L_]  = (__bf16)r1;
}

// ------------------------------ generic GEMM -------------------------------
// C[M,N] = A[M,K](bf16,rowmajor) * W[N,K](bf16,rowmajor)^T (+bias).
// 256 threads, 64x64 block tile, 16x16x32 WMMA, async-LDS staged.
// row_mode==1 remaps output row r=l*B+b -> (b*L + l) (energy layout).

#define BM 64
#define BN 64
#define BK 32

__global__ __launch_bounds__(256)
void k_gemm(const __bf16* __restrict__ Ag, const __bf16* __restrict__ Wg,
            const float* __restrict__ bias, float* Cf, __bf16* Cbf,
            int M, int N, int K, int ldc, int col_off, int row_mode,
            long bsA, long bsW, long bsC) {
  __shared__ __align__(32) __bf16 As[BM][BK];
  __shared__ __align__(32) __bf16 Bs[BN][BK];
  int bz = blockIdx.z;
  Ag += (long)bz * bsA;
  Wg += (long)bz * bsW;
  long coff = (long)bz * bsC;
  int bm0 = blockIdx.y * BM, bn0 = blockIdx.x * BN;
  int tid = threadIdx.x, w = tid >> 5, lane = tid & 31;
  int mlane = lane & 15, hi = lane >> 4;
  int mi = w & 3, ni0 = w >> 2;                 // 2 tiles/wave: ni0, ni0+2
  int row_ld = tid >> 2, part = tid & 3;
  v8f acc0 = {}, acc1 = {};
  for (int k0 = 0; k0 < K; k0 += BK) {
    __syncthreads();
    // gfx1250 async path: memory -> LDS without VGPR round-trip (ASYNCcnt)
    async_g2l_b128(&As[row_ld][part * 8],
                   &Ag[(long)(bm0 + row_ld) * K + k0 + part * 8]);
    async_g2l_b128(&Bs[row_ld][part * 8],
                   &Wg[(long)(bn0 + row_ld) * K + k0 + part * 8]);
    if (k0 + BK < K) {   // hint next tiles into cache (global_prefetch)
      __builtin_prefetch(
          (const void*)&Ag[(long)(bm0 + row_ld) * K + k0 + BK + part * 8]);
      __builtin_prefetch(
          (const void*)&Wg[(long)(bn0 + row_ld) * K + k0 + BK + part * 8]);
    }
    wait_asynccnt0();
    __syncthreads();
    union { v16bf v; unsigned u[8]; } af;
    int arow = mi * 16 + mlane;
#pragma unroll
    for (int vv = 0; vv < 8; ++vv) {
      int kk = hi * 16 + vv * 2;
      af.u[vv] = *(const unsigned*)&As[arow][kk];
    }
    v16bf b0 = *(const v16bf*)&Bs[ni0 * 16 + mlane][hi * 16];
    v16bf b1 = *(const v16bf*)&Bs[(ni0 + 2) * 16 + mlane][hi * 16];
    acc0 = wmma_bf16(af.v, b0, acc0);
    acc1 = wmma_bf16(af.v, b1, acc1);
  }
#pragma unroll
  for (int r = 0; r < 8; ++r) {
    int mrow = bm0 + mi * 16 + r + hi * 8;
#pragma unroll
    for (int which = 0; which < 2; ++which) {
      int ncol = bn0 + (ni0 + 2 * which) * 16 + mlane;
      float vv = which ? acc1[r] : acc0[r];
      if (bias) vv += bias[ncol];
      long orow = (row_mode == 1)
                      ? ((long)(mrow & (B_ - 1)) * L_ + (mrow >> 5))
                      : (long)mrow;
      long oidx = coff + orow * (long)ldc + col_off + ncol;
      if (Cf)  Cf[oidx]  = vv;
      if (Cbf) Cbf[oidx] = (__bf16)vv;
    }
  }
}

// ------------------------------ BiGRU scan ---------------------------------
// One workgroup per direction (grid.x = 2), 1024 threads = 32 waves.
// Per step: gh = h(32x256) @ whh^T(256x768) via 96 WMMA tiles (3/wave),
// then gate math in f32.  h kept f32 (recurrence) + bf16 mirror (WMMA A-frag)
// in LDS.  Dynamic LDS: h f32 32KB + gh f32 96KB + h bf16 16KB = 144 KB.

__global__ __launch_bounds__(1024)
void k_gru_scan(const float* __restrict__ giF, const float* __restrict__ giB,
                const __bf16* __restrict__ whh2,   // (2,768,256) bf16
                const float* __restrict__ bhh2,    // (2,768)
                const float* __restrict__ h0,      // (2,B,H)
                __bf16* __restrict__ out) {        // (L,B,2H) bf16
  extern __shared__ float dynsm[];   // [0,8192)=hF [8192,32768)=gh [32768..)=hB
  __bf16* hB = (__bf16*)(dynsm + 32768);
  const int dir = blockIdx.x;
  const float* gi = dir ? giB : giF;
  const __bf16* whh = whh2 + (size_t)dir * G3_ * H_;
  const float* bhh = bhh2 + dir * G3_;
  const int tid = threadIdx.x;
  for (int i = tid; i < B_ * H_; i += 1024) {
    float h = h0[dir * B_ * H_ + i];
    dynsm[i] = h;
    hB[i] = (__bf16)h;
  }
  __syncthreads();
  const int w = tid >> 5, lane = tid & 31;
  const int mlane = lane & 15, hi = lane >> 4;
  const int tile0 = w * 3;
  const int mi = tile0 / 48;          // same for all 3 tiles of this wave
  const int ni0 = tile0 % 48;
  const int arow = mi * 16 + mlane;

  for (int t = 0; t < L_; ++t) {
    const int tt = dir ? (L_ - 1 - t) : t;
    v8f acc0 = {}, acc1 = {}, acc2 = {};
#pragma unroll
    for (int kk = 0; kk < H_ / 32; ++kk) {
      const int kbase = kk * 32;
      union { v16bf v; unsigned u[8]; } af;
#pragma unroll
      for (int vv = 0; vv < 8; ++vv) {
        int k0 = kbase + hi * 16 + vv * 2;
        af.u[vv] = *(const unsigned*)&hB[arow * H_ + k0];
      }
      const __bf16* wp = whh + kbase + hi * 16;
      v16bf b0 = *(const v16bf*)(wp + (size_t)((ni0 + 0) * 16 + mlane) * H_);
      v16bf b1 = *(const v16bf*)(wp + (size_t)((ni0 + 1) * 16 + mlane) * H_);
      v16bf b2 = *(const v16bf*)(wp + (size_t)((ni0 + 2) * 16 + mlane) * H_);
      acc0 = wmma_bf16(af.v, b0, acc0);
      acc1 = wmma_bf16(af.v, b1, acc1);
      acc2 = wmma_bf16(af.v, b2, acc2);
    }
#pragma unroll
    for (int r = 0; r < 8; ++r) {
      int m = mi * 16 + r + hi * 8;
      dynsm[8192 + m * G3_ + (ni0 + 0) * 16 + mlane] = acc0[r];
      dynsm[8192 + m * G3_ + (ni0 + 1) * 16 + mlane] = acc1[r];
      dynsm[8192 + m * G3_ + (ni0 + 2) * 16 + mlane] = acc2[r];
    }
    __syncthreads();
#pragma unroll
    for (int i = 0; i < 8; ++i) {
      int e = tid * 8 + i;                        // 8192 gate columns
      int b = e >> 8, j = e & 255;
      const float* gr = gi + ((size_t)tt * B_ + b) * G3_;
      float i_r = gr[j], i_z = gr[H_ + j], i_n = gr[2 * H_ + j];
      float h_r = dynsm[8192 + b * G3_ + j]          + bhh[j];
      float h_z = dynsm[8192 + b * G3_ + H_ + j]     + bhh[H_ + j];
      float h_n = dynsm[8192 + b * G3_ + 2 * H_ + j] + bhh[2 * H_ + j];
      float r = 1.0f / (1.0f + expf(-(i_r + h_r)));
      float z = 1.0f / (1.0f + expf(-(i_z + h_z)));
      float n = tanhf(i_n + r * h_n);
      float hp = dynsm[b * H_ + j];
      float hn = (1.0f - z) * n + z * hp;
      dynsm[b * H_ + j] = hn;
      hB[b * H_ + j] = (__bf16)hn;
      out[((size_t)tt * B_ + b) * H2_ + dir * H_ + j] = (__bf16)hn;
    }
    __syncthreads();
  }
}

// ------------------------------- host side ---------------------------------

extern "C" void kernel_launch(void* const* d_in, const int* in_sizes, int n_in,
                              void* d_out, int out_size, void* d_ws,
                              size_t ws_size, hipStream_t stream) {
  const int*   tok    = (const int*)d_in[0];
  const float* hidden = (const float*)d_in[2];
  const float* feat   = (const float*)d_in[3];
  const float* embT   = (const float*)d_in[4];
  const float* wih0   = (const float*)d_in[5];
  const float* whh0   = (const float*)d_in[6];
  const float* bih0   = (const float*)d_in[7];
  const float* bhh0   = (const float*)d_in[8];
  const float* wih1   = (const float*)d_in[9];
  const float* whh1   = (const float*)d_in[10];
  const float* bih1   = (const float*)d_in[11];
  const float* bhh1   = (const float*)d_in[12];
  const float* la_w   = (const float*)d_in[13];
  const float* la_b   = (const float*)d_in[14];
  const float* fa_w   = (const float*)d_in[15];
  const float* fa_b   = (const float*)d_in[16];
  const float* vmat   = (const float*)d_in[17];

  float* out_ctx = (float*)d_out;                           // (B,F,2H)
  float* out_wts = out_ctx + (size_t)B_ * F_ * H2_;         // (B,F,L)

  // ---- workspace carve-out (bytes, 256B aligned) ----
  char* base = (char*)d_ws;
  size_t off = 0;
  auto carve = [&](size_t bytes) {
    char* p = base + off;
    off = (off + bytes + 255) & ~(size_t)255;
    return p;
  };
  __bf16* wih0_bf = (__bf16*)carve(2L * G3_ * E_ * 2);
  __bf16* whh0_bf = (__bf16*)carve(2L * G3_ * H_ * 2);
  __bf16* wih1_bf = (__bf16*)carve(2L * G3_ * H2_ * 2);
  __bf16* whh1_bf = (__bf16*)carve(2L * G3_ * H_ * 2);
  __bf16* law_bf  = (__bf16*)carve((size_t)H_ * H2_ * 2);
  __bf16* faw_bf  = (__bf16*)carve((size_t)E_ * H2_ * 2);
  __bf16* feat_bf = (__bf16*)carve((size_t)NF_ * E_ * 2);
  __bf16* v_bf    = (__bf16*)carve((size_t)PAD_ * H_ * 2);
  __bf16* x_bf    = (__bf16*)carve((size_t)L_ * B_ * E_ * 2);
  float*  gi_f    = (float*)carve((size_t)L_ * B_ * G3_ * 4);
  float*  gi_b    = (float*)carve((size_t)L_ * B_ * G3_ * 4);
  __bf16* out0_bf = (__bf16*)carve((size_t)L_ * B_ * H2_ * 2);
  __bf16* rnn_bf  = (__bf16*)carve((size_t)L_ * B_ * H2_ * 2);
  __bf16* wh_bf   = (__bf16*)carve((size_t)L_ * B_ * H_ * 2);
  __bf16* wh2_bf  = (__bf16*)carve((size_t)L_ * B_ * E_ * 2);
  float*  energy  = (float*)carve((size_t)B_ * L_ * F_ * 4);
  __bf16* wat_bf  = (__bf16*)carve((size_t)B_ * F_ * L_ * 2);
  __bf16* rnnT_bf = (__bf16*)carve((size_t)B_ * H2_ * L_ * 2);
  (void)ws_size; (void)n_in; (void)in_sizes; (void)out_size;

  auto cvt = [&](const float* s, __bf16* d, long n) {
    k_f32_to_bf16<<<(unsigned)((n + 255) / 256), 256, 0, stream>>>(s, d, n);
  };
  cvt(wih0, wih0_bf, 2L * G3_ * E_);
  cvt(whh0, whh0_bf, 2L * G3_ * H_);
  cvt(wih1, wih1_bf, 2L * G3_ * H2_);
  cvt(whh1, whh1_bf, 2L * G3_ * H_);
  cvt(la_w, law_bf, (long)H_ * H2_);
  cvt(fa_w, faw_bf, (long)E_ * H2_);
  cvt(feat, feat_bf, (long)NF_ * E_);
  cvt(vmat, v_bf, (long)PAD_ * H_);

  // embedding gather
  {
    long n = (long)L_ * B_ * E_;
    k_embed<<<(unsigned)((n + 255) / 256), 256, 0, stream>>>(tok, embT, x_bf);
  }

  const int M = L_ * B_;   // 16384
  auto gemm = [&](const __bf16* A, const __bf16* W, const float* bias, float* Cf,
                  __bf16* Cbf, int m, int n, int k, int ldc, int col_off,
                  int row_mode, long bsA, long bsW, long bsC, int batches) {
    dim3 grid(n / BN, m / BM, batches);
    k_gemm<<<grid, 256, 0, stream>>>(A, W, bias, Cf, Cbf, m, n, k, ldc, col_off,
                                     row_mode, bsA, bsW, bsC);
  };

  // allow 144KB dynamic LDS for the GRU scan
  (void)hipFuncSetAttribute((const void*)k_gru_scan,
                            hipFuncAttributeMaxDynamicSharedMemorySize,
                            147456);

  // ---- layer 0 ----
  gemm(x_bf, wih0_bf,            bih0,        gi_f, nullptr, M, G3_, E_, G3_, 0, 0, 0, 0, 0, 1);
  gemm(x_bf, wih0_bf + G3_ * E_, bih0 + G3_,  gi_b, nullptr, M, G3_, E_, G3_, 0, 0, 0, 0, 0, 1);
  k_gru_scan<<<2, 1024, 147456, stream>>>(gi_f, gi_b, whh0_bf, bhh0, hidden,
                                          out0_bf);
  // ---- layer 1 ----
  gemm(out0_bf, wih1_bf,             bih1,       gi_f, nullptr, M, G3_, H2_, G3_, 0, 0, 0, 0, 0, 1);
  gemm(out0_bf, wih1_bf + G3_ * H2_, bih1 + G3_, gi_b, nullptr, M, G3_, H2_, G3_, 0, 0, 0, 0, 0, 1);
  k_gru_scan<<<2, 1024, 147456, stream>>>(gi_f, gi_b, whh1_bf, bhh1,
                                          hidden + 2L * B_ * H_, rnn_bf);

  // ---- attention energies ----
  gemm(rnn_bf, law_bf, la_b, nullptr, wh_bf,  M, H_, H2_, H_, 0, 0, 0, 0, 0, 1);
  gemm(rnn_bf, faw_bf, fa_b, nullptr, wh2_bf, M, E_, H2_, E_, 0, 0, 0, 0, 0, 1);
  gemm(wh2_bf, feat_bf, nullptr, energy, nullptr, M, NF_,  E_, F_, 0,   1, 0, 0, 0, 1);
  gemm(wh_bf,  v_bf,    nullptr, energy, nullptr, M, PAD_, H_, F_, NF_, 1, 0, 0, 0, 1);

  // ---- softmax -> attn_weights (f32 out + bf16 scratch) ----
  k_softmax<<<B_ * L_, 256, 0, stream>>>(energy, out_wts, wat_bf);

  // ---- context = weights @ rnn  (batched over B) ----
  {
    long n = (long)B_ * H2_ * L_;
    k_transpose_rnn<<<(unsigned)((n + 255) / 256), 256, 0, stream>>>(rnn_bf,
                                                                     rnnT_bf);
  }
  gemm(wat_bf, rnnT_bf, nullptr, out_ctx, nullptr, F_, H2_, L_, H2_, 0, 0,
       (long)F_ * L_, (long)H2_ * L_, (long)F_ * H2_, B_);
}
